// CompSAE_34711925686336
// MI455X (gfx1250) — matrix operand order repaired and checked by reading, not compiled, not run on hardware
//
#include <hip/hip_runtime.h>
#include <hip/hip_bf16.h>

// ---------------------------------------------------------------------------
// Fused CompSAE forward for gfx1250 (MI455X).
//   f = relu(x @ W_enc_f + b_eff)   where b_eff = b_enc_f - b_dec @ W_enc_f
//   c = relu(f @ W_enc_c + b_enc_c)                  [N,128]   K=256
//   t = relu(c @ B2 + b2)       (attention collapsed)[N,32]    K=128
//   out = t @ W_dec + f @ W_res + b_dec              [N,2048]
// One pass over acts, one pass writing out; f/c/t live in LDS only.
// Stage-1 weights are staged per-chunk into LDS with
// global_load_async_to_lds_b128 (ASYNCcnt), double-buffered, published with
// a fence-light split barrier (asynccnt/dscnt only -> acts loads stay in
// flight across the barrier). All matmuls via v_wmma_f32_16x16x32_bf16.
// Block = 256 threads = 8 waves over 64 rows: 4 row-groups x 2 col-halves.
// ---------------------------------------------------------------------------

typedef __attribute__((ext_vector_type(16))) __bf16 v16bf;
typedef __attribute__((ext_vector_type(8)))  float  v8f;

#define D_IN  2048
#define D_F   256
#define D_C   128
#define KBOT  32
#define ROWS_PER_BLOCK 64
#define THREADS 256

// LDS arena overlay:
//   during stage-1 k-loop : [0 .. 40959]  two 20 KB async weight buffers
//                           (256 cols x 80 B: 64 B chunk data + 16 B pad)
//   after stage-1 loop    : f (32 KB) | c (16 KB) | t (4 KB)
#define WBUF_STRIDE 80
#define WBUF_BYTES  (256 * WBUF_STRIDE)      // 20480 per buffer
#define ARENA_BYTES 53248                    // max(2*20480, 32K+16K+4K)

__device__ __forceinline__ v8f vzero8() {
    v8f z;
#pragma unroll
    for (int i = 0; i < 8; ++i) z[i] = 0.0f;
    return z;
}

__device__ __forceinline__ v8f wmma_bf16(v16bf a, v16bf b, v8f c) {
    // (neg_a, A, neg_b, B, c_mod, C, reuse_a, reuse_b)
    return __builtin_amdgcn_wmma_f32_16x16x32_bf16(false, a, false, b,
                                                   (short)0, c, false, false);
}

// Fence-light pipeline barrier for the async double buffer:
// only drains this wave's async-to-LDS writes and LDS reads; global loads
// (read-only acts stream) are allowed to stay in flight across the barrier.
__device__ __forceinline__ void pipeline_barrier() {
    asm volatile(
        "s_wait_asynccnt 0x0\n\t"
        "s_wait_dscnt 0x0\n\t"
        "s_barrier_signal -1\n\t"
        "s_barrier_wait -1"
        ::: "memory");
}

// A-fragment (16x32 bf16) for this lane from an f32 row (convert only).
// element i <-> k = kb + 16*hi + i
__device__ __forceinline__ v16bf load_a_f32(const float* __restrict__ row,
                                            int kb, int hi) {
    v16bf a;
    const int k0 = kb + hi * 16;
#pragma unroll
    for (int i = 0; i < 8; ++i) a[i] = (__bf16)row[k0 + i];
    const int k1 = kb + hi * 16 + 8;
#pragma unroll
    for (int i = 0; i < 8; ++i) a[i + 8] = (__bf16)row[k1 + i];
    return a;
}

// A-fragment from a bf16 row in LDS.
__device__ __forceinline__ v16bf load_a_lds(const __bf16* row, int kb, int hi) {
    v16bf a;
    const int k0 = kb + hi * 16;
#pragma unroll
    for (int i = 0; i < 8; ++i) a[i] = row[k0 + i];
    const int k1 = kb + hi * 16 + 8;
#pragma unroll
    for (int i = 0; i < 8; ++i) a[i + 8] = row[k1 + i];
    return a;
}

// ---------------------------------------------------------------------------
// Prologue: f32 [K][N] -> bf16 [N][K]  (B-fragment-friendly layout)
// ---------------------------------------------------------------------------
__global__ void k_transpose_bf16(const float* __restrict__ src,
                                 __bf16* __restrict__ dst,
                                 int K, int Ncols) {
    int idx = blockIdx.x * blockDim.x + threadIdx.x;
    if (idx >= K * Ncols) return;
    int n = idx / K;
    int k = idx - n * K;
    dst[idx] = (__bf16)src[(size_t)k * Ncols + n];
}

// ---------------------------------------------------------------------------
// Prologue: fold the pre-subtraction of b_dec into the encoder bias:
//   b_eff[col] = b_enc_f[col] - sum_k b_dec[k] * W_enc_f[k][col]
// ---------------------------------------------------------------------------
__global__ void k_bias_fold(const float* __restrict__ b_dec,
                            const float* __restrict__ W_enc_f,
                            const float* __restrict__ b_enc_f,
                            float* __restrict__ b_eff) {
    int col = blockIdx.x * blockDim.x + threadIdx.x;
    if (col >= D_F) return;
    float s = b_enc_f[col];
    for (int k = 0; k < D_IN; ++k) s -= b_dec[k] * W_enc_f[(size_t)k * D_F + col];
    b_eff[col] = s;
}

// ---------------------------------------------------------------------------
// Prologue: collapse value/out projections + bottleneck into B2 [128x32]
//   M[k][j]  = sum_t W_v[t][k] * W_out[j][t]          (= (W_out W_v)^T)
//   B2[k][n] = sum_j M[k][j] * W_b[j][n]
//   b2[n]    = sum_j (b_out[j] + sum_t W_out[j][t] b_v[t]) * W_b[j][n]
// Stored transposed/bf16: B2_t[n*128 + k].
// ---------------------------------------------------------------------------
__global__ void k_attn_collapse(const float* __restrict__ W_v,
                                const float* __restrict__ b_v,
                                const float* __restrict__ W_out,
                                const float* __restrict__ b_out,
                                const float* __restrict__ W_b,
                                __bf16* __restrict__ B2_t,
                                float* __restrict__ b2) {
    __shared__ __bf16 Ms[D_C * D_C];   // 32 KB
    __shared__ float  bias1[D_C];
    const int tid = threadIdx.x;       // 256 threads

    for (int idx = tid; idx < D_C * D_C; idx += 256) {
        int k = idx >> 7, j = idx & 127;
        float s = 0.0f;
        for (int t = 0; t < D_C; ++t)
            s += W_v[t * D_C + k] * W_out[j * D_C + t];
        Ms[idx] = (__bf16)s;
    }
    for (int j = tid; j < D_C; j += 256) {
        float s = b_out[j];
        for (int t = 0; t < D_C; ++t) s += W_out[j * D_C + t] * b_v[t];
        bias1[j] = s;
    }
    __syncthreads();
    for (int idx = tid; idx < KBOT * D_C; idx += 256) {
        int n = idx >> 7, k = idx & 127;
        float s = 0.0f;
        for (int j = 0; j < D_C; ++j)
            s += (float)Ms[k * D_C + j] * W_b[j * KBOT + n];
        B2_t[n * D_C + k] = (__bf16)s;
    }
    for (int n = tid; n < KBOT; n += 256) {
        float s = 0.0f;
        for (int j = 0; j < D_C; ++j) s += bias1[j] * W_b[j * KBOT + n];
        b2[n] = s;
    }
}

// ---------------------------------------------------------------------------
// Main fused kernel. 256 threads = 8 waves; wave = (rowGroup g, colHalf h).
// Each wave owns 16 rows x half the columns of every stage.
// ---------------------------------------------------------------------------
__global__ __launch_bounds__(THREADS, 1)
void k_sae_fused(const float* __restrict__ acts,
                 const __bf16* __restrict__ Wf_t,    // [256][2048]
                 const float* __restrict__ b_eff,
                 const __bf16* __restrict__ Wc_t,    // [128][256]
                 const float* __restrict__ b_enc_c,
                 const __bf16* __restrict__ B2_t,    // [32][128]
                 const float* __restrict__ b2,
                 const __bf16* __restrict__ Wdec_t,  // [2048][32]
                 const __bf16* __restrict__ Wres_t,  // [2048][256]
                 const float* __restrict__ b_dec,
                 float* __restrict__ out) {
    __shared__ char  arena[ARENA_BYTES] __attribute__((aligned(128)));
    __shared__ float bdec_lds[D_IN];                 // 8 KB

    __bf16* f_lds = (__bf16*)arena;                  // 32 KB (after stage 1)
    __bf16* c_lds = (__bf16*)(arena + 32768);        // 16 KB
    __bf16* t_lds = (__bf16*)(arena + 49152);        //  4 KB

    const int tid  = threadIdx.x;
    const int wave = tid >> 5;
    const int lane = tid & 31;
    const int ln   = lane & 15;   // row (A frag) or col (B/D frag) index
    const int hi   = lane >> 4;   // upper half-wave selector
    const int g    = wave & 3;    // row group (16 rows each)
    const int h    = wave >> 2;   // column half

    for (int i = tid; i < D_IN; i += THREADS) bdec_lds[i] = b_dec[i];
    __syncthreads();

    const int  rowL = g * 16;
    const long rowG = (long)blockIdx.x * ROWS_PER_BLOCK + rowL;
    const float* actRow = acts + (rowG + ln) * (long)D_IN;

    // ---------------- stage 1: f = relu(x @ W_enc_f + b_eff) ---------------
    // Weight chunk (256 cols x 32 k) staged to LDS via async-to-LDS copies,
    // double-buffered; published with s_wait_asynccnt + split barrier.
    v8f accF[8];
#pragma unroll
    for (int t = 0; t < 8; ++t) accF[t] = vzero8();

    const int colT = tid;  // each thread stages one weight column
    const __bf16* gcol = Wf_t + (size_t)colT * D_IN;
    const uint32_t arenaOff = (uint32_t)(uintptr_t)(void*)arena;

    auto stage_chunk = [&](int kb, int buf) {
        const uint32_t l0 = arenaOff + (uint32_t)(buf * WBUF_BYTES +
                                                  colT * WBUF_STRIDE);
        const __bf16* g0 = gcol + kb;
#pragma unroll
        for (int p = 0; p < 4; ++p) {
            asm volatile("global_load_async_to_lds_b128 %0, %1, off"
                         :: "v"(l0 + p * 16), "v"(g0 + p * 8)
                         : "memory");
        }
    };

    stage_chunk(0, 0);
    v16bf aC = load_a_f32(actRow, 0, hi);
    pipeline_barrier();                      // publish chunk 0

    int cur = 0;
    for (int kb = 0; kb < D_IN; kb += 32) {
        if (kb + 32 < D_IN) stage_chunk(kb + 32, cur ^ 1);
        __builtin_prefetch(actRow + ((kb + 64 < D_IN) ? kb + 64 : 0), 0, 3);
        const int kn = (kb + 32 < D_IN) ? kb + 32 : 0;
        v16bf aN = load_a_f32(actRow, kn, hi);

        // Preload ALL B fragments of this chunk (16 ds_load_b128 in flight),
        // then issue the 8 WMMAs -> partial dscnt waits instead of per-tile
        // full drains.
        const char* wb = arena + cur * WBUF_BYTES;
        v16bf bfr[8];
#pragma unroll
        for (int nt = 0; nt < 8; ++nt) {
            const int col = h * 128 + nt * 16 + ln;
            bfr[nt] = *(const v16bf*)(wb + col * WBUF_STRIDE + hi * 32);
        }
#pragma unroll
        for (int nt = 0; nt < 8; ++nt)
            accF[nt] = wmma_bf16(aC, bfr[nt], accF[nt]);
        aC = aN;

        pipeline_barrier();                  // publish next chunk / retire cur
        cur ^= 1;
    }
    // arena is now reused as f/c/t storage (loop-final barrier retired wbuf).
#pragma unroll
    for (int nt = 0; nt < 8; ++nt) {
        const int col  = h * 128 + nt * 16 + ln;
        const float bb = b_eff[col];
#pragma unroll
        for (int j = 0; j < 8; ++j) {
            float v = accF[nt][j] + bb;
            v = v > 0.0f ? v : 0.0f;
            f_lds[(rowL + hi * 8 + j) * D_F + col] = (__bf16)v;
        }
    }
    __syncthreads();

    // ---------------- stage 2: c = relu(f @ W_enc_c + b_enc_c) -------------
    // wave covers col tiles [h*4 .. h*4+3]; register double-buffered B.
    const __bf16* fRow = f_lds + (rowL + ln) * D_F;
    v8f accC[4];
#pragma unroll
    for (int t = 0; t < 4; ++t) accC[t] = vzero8();

    const __bf16* wcp = Wc_t + (size_t)(h * 64 + ln) * D_F + hi * 16;
    v16bf c0[4], c1[4];
#pragma unroll
    for (int nt = 0; nt < 4; ++nt)
        c0[nt] = *(const v16bf*)(wcp + (size_t)nt * 16 * D_F);

    for (int kb = 0; kb < D_F; kb += 64) {
        v16bf a0 = load_a_lds(fRow, kb, hi);
#pragma unroll
        for (int nt = 0; nt < 4; ++nt)
            c1[nt] = *(const v16bf*)(wcp + (size_t)nt * 16 * D_F + kb + 32);
#pragma unroll
        for (int nt = 0; nt < 4; ++nt)
            accC[nt] = wmma_bf16(a0, c0[nt], accC[nt]);

        v16bf a1 = load_a_lds(fRow, kb + 32, hi);
        const int kn = (kb + 64 < D_F) ? kb + 64 : 0;
#pragma unroll
        for (int nt = 0; nt < 4; ++nt)
            c0[nt] = *(const v16bf*)(wcp + (size_t)nt * 16 * D_F + kn);
#pragma unroll
        for (int nt = 0; nt < 4; ++nt)
            accC[nt] = wmma_bf16(a1, c1[nt], accC[nt]);
    }
#pragma unroll
    for (int nt = 0; nt < 4; ++nt) {
        const int col  = h * 64 + nt * 16 + ln;
        const float bb = b_enc_c[col];
#pragma unroll
        for (int j = 0; j < 8; ++j) {
            float v = accC[nt][j] + bb;
            v = v > 0.0f ? v : 0.0f;
            c_lds[(rowL + hi * 8 + j) * D_C + col] = (__bf16)v;
        }
    }
    __syncthreads();

    // ---------------- stage 3: t = relu(c @ B2 + b2) -----------------------
    // wave covers col tile h (cols h*16 .. h*16+15).
    const __bf16* cRow = c_lds + (rowL + ln) * D_C;
    v8f accT = vzero8();
    const __bf16* bp = B2_t + (size_t)(h * 16 + ln) * D_C + hi * 16;
#pragma unroll
    for (int kb = 0; kb < D_C; kb += 32) {
        v16bf a = load_a_lds(cRow, kb, hi);
        v16bf b = *(const v16bf*)(bp + kb);
        accT = wmma_bf16(a, b, accT);
    }
    {
        const int col  = h * 16 + ln;
        const float bb = b2[col];
#pragma unroll
        for (int j = 0; j < 8; ++j) {
            float v = accT[j] + bb;
            v = v > 0.0f ? v : 0.0f;
            t_lds[(rowL + hi * 8 + j) * KBOT + col] = (__bf16)v;
        }
    }
    __syncthreads();

    // ---------------- stage 4: out = t @ W_dec + f @ W_res + b_dec ---------
    // wave covers col tiles [h*64 .. h*64+63]; two-deep B double-buffer.
    const __bf16* tRow = t_lds + (rowL + ln) * KBOT;
    v16bf tA = load_a_lds(tRow, 0, hi);        // whole K=32 of t, hoisted
    v16bf fA[8];                               // whole K=256 of f, hoisted
#pragma unroll
    for (int kc = 0; kc < 8; ++kc) fA[kc] = load_a_lds(fRow, kc * 32, hi);

    const __bf16* wdp = Wdec_t + (size_t)(h * 1024 + ln) * KBOT + hi * 16;
    const __bf16* wrp = Wres_t + (size_t)(h * 1024 + ln) * D_F + hi * 16;

    v16bf bd0, bd1, br0[8], br1[8];
    bd0 = *(const v16bf*)(wdp);
#pragma unroll
    for (int kc = 0; kc < 8; ++kc)
        br0[kc] = *(const v16bf*)(wrp + kc * 32);

    auto emit_tile = [&](int nt, v16bf bd, const v16bf* br) {
        v8f acc = wmma_bf16(tA, bd, vzero8());
#pragma unroll
        for (int kc = 0; kc < 8; ++kc)
            acc = wmma_bf16(fA[kc], br[kc], acc);
        const int col = h * 1024 + nt * 16 + ln;
        const float bdv = bdec_lds[col];
#pragma unroll
        for (int j = 0; j < 8; ++j)
            out[(rowG + hi * 8 + j) * (long)D_IN + col] = acc[j] + bdv;
    };

    for (int nt = 0; nt < 64; nt += 2) {
        // prefetch tile nt+1 into bank 1, then emit tile nt from bank 0
        bd1 = *(const v16bf*)(wdp + (size_t)(nt + 1) * 16 * KBOT);
#pragma unroll
        for (int kc = 0; kc < 8; ++kc)
            br1[kc] = *(const v16bf*)(wrp + (size_t)(nt + 1) * 16 * D_F + kc * 32);
        emit_tile(nt, bd0, br0);

        // prefetch tile nt+2 into bank 0, then emit tile nt+1 from bank 1
        const int np = (nt + 2 < 64) ? nt + 2 : 0;
        bd0 = *(const v16bf*)(wdp + (size_t)np * 16 * KBOT);
#pragma unroll
        for (int kc = 0; kc < 8; ++kc)
            br0[kc] = *(const v16bf*)(wrp + (size_t)np * 16 * D_F + kc * 32);
        emit_tile(nt + 1, bd1, br1);
    }
}

// ---------------------------------------------------------------------------
// Workspace layout (bytes, all 64B-aligned):
//   Wf_t   [256][2048] bf16 @ 0         (1 MB)
//   Wres_t [2048][256] bf16 @ 1048576   (1 MB)
//   Wc_t   [128][256]  bf16 @ 2097152   (64 KB)
//   Wdec_t [2048][32]  bf16 @ 2162688   (128 KB)
//   B2_t   [32][128]   bf16 @ 2293760   (8 KB)
//   b2     [32]        f32  @ 2301952   (128 B)
//   b_eff  [256]       f32  @ 2302080   (1 KB)
// ---------------------------------------------------------------------------
extern "C" void kernel_launch(void* const* d_in, const int* in_sizes, int n_in,
                              void* d_out, int out_size, void* d_ws, size_t ws_size,
                              hipStream_t stream) {
    const float* acts    = (const float*)d_in[0];
    const float* W_enc_f = (const float*)d_in[1];
    const float* b_enc_f = (const float*)d_in[2];
    const float* W_enc_c = (const float*)d_in[3];
    const float* b_enc_c = (const float*)d_in[4];
    const float* W_v     = (const float*)d_in[5];
    const float* b_v     = (const float*)d_in[6];
    const float* W_out   = (const float*)d_in[7];
    const float* b_out   = (const float*)d_in[8];
    const float* W_bot   = (const float*)d_in[9];
    const float* W_dec   = (const float*)d_in[10];
    const float* b_dec   = (const float*)d_in[11];
    const float* W_res   = (const float*)d_in[12];
    float* out = (float*)d_out;

    char* ws = (char*)d_ws;
    __bf16* Wf_t   = (__bf16*)(ws + 0);
    __bf16* Wres_t = (__bf16*)(ws + 1048576);
    __bf16* Wc_t   = (__bf16*)(ws + 2097152);
    __bf16* Wdec_t = (__bf16*)(ws + 2162688);
    __bf16* B2_t   = (__bf16*)(ws + 2293760);
    float*  b2     = (float*) (ws + 2301952);
    float*  b_eff  = (float*) (ws + 2302080);

    k_transpose_bf16<<<(D_IN * D_F + 255) / 256, 256, 0, stream>>>(W_enc_f, Wf_t, D_IN, D_F);
    k_transpose_bf16<<<(D_F * D_IN + 255) / 256, 256, 0, stream>>>(W_res, Wres_t, D_F, D_IN);
    k_transpose_bf16<<<(D_F * D_C + 255) / 256, 256, 0, stream>>>(W_enc_c, Wc_t, D_F, D_C);
    k_transpose_bf16<<<(KBOT * D_IN + 255) / 256, 256, 0, stream>>>(W_dec, Wdec_t, KBOT, D_IN);
    k_bias_fold<<<1, 256, 0, stream>>>(b_dec, W_enc_f, b_enc_f, b_eff);
    k_attn_collapse<<<1, 256, 0, stream>>>(W_v, b_v, W_out, b_out, W_bot, B2_t, b2);

    const int N = in_sizes[0] / D_IN;  // 131072
    k_sae_fused<<<N / ROWS_PER_BLOCK, THREADS, 0, stream>>>(
        acts, Wf_t, b_eff, Wc_t, b_enc_c, B2_t, b2, Wdec_t, Wres_t, b_dec, out);
}
